// Block3D_22256520528391
// MI455X (gfx1250) — matrix operand order repaired and checked by reading, not compiled, not run on hardware
//
#include <hip/hip_runtime.h>
#include <hip/hip_bf16.h>

typedef __attribute__((ext_vector_type(16))) _Float16 v16h;
typedef __attribute__((ext_vector_type(8)))  float    v8f;

#define NPOS 32768          // 32*32*32
#define GN_EPS 1e-5f

// ---------------------------------------------------------------------------
// B-matrix 32x16 f16 fragment from f32 activations (contiguous k order chosen here, matching the A side):
// lane L holds column N = L&15; lanes 0-15: elems -> K=0..15 ; lanes 16-31: K=16..31
// Each of the 16 loads is a fully coalesced 128B request across the wave.
// ---------------------------------------------------------------------------
__device__ __forceinline__ v16h load_b_frag(const float* __restrict__ X,
                                            int ldx, int k0, int n0, int lane) {
    int n  = n0 + (lane & 15);
    int kb = (lane & 16) ? 16 : 0;
    const float* col = X + (size_t)(k0 + kb) * ldx + n;
    v16h b;
#pragma unroll
    for (int e = 0; e < 16; ++e) b[e] = (_Float16)col[(size_t)e * ldx];
    return b;
}

// ---------------------------------------------------------------------------
// 1x1x1 conv as channel GEMM:  Y[b,o,n] = act(sc[o]*(W@X) + bi[o]) (+resid)
// One wave owns a 16-column strip over ALL output channels:
//   - W staged to LDS as f16 once per block (A-side traffic -> ds path)
//   - B fragment loaded once per K-step, reused by all MT m-tiles
// A-fragment lane layout: row M = lane&15; lanes<16: K=kb+{0..15} kb=0,
// lanes>=16: kb=16 -> two contiguous, 16B-aligned 8-half runs in LDS.
// ---------------------------------------------------------------------------
template<int O, int I, bool HAS_SCALE, bool RELU, bool RESID>
__global__ __launch_bounds__(256)
void conv1x1_wmma(const float* __restrict__ X, const float* __restrict__ W,
                  const float* __restrict__ sc, const float* __restrict__ bi,
                  const float* __restrict__ resid, float* __restrict__ Y) {
    constexpr int MT = (O + 15) / 16;
    __shared__ _Float16 Wl[O * I];
    for (int i = threadIdx.x; i < O * I; i += 256) Wl[i] = (_Float16)W[i];
    __syncthreads();

    const int NT = NPOS / 16;              // 2048 n-tiles per batch
    int wid  = blockIdx.x * 8 + (threadIdx.x >> 5);
    int lane = threadIdx.x & 31;
    int nt = wid % NT;
    int b  = wid / NT;
    int n0 = nt * 16;

    const float* Xb = X + (size_t)b * I * NPOS;
    v8f zero = {0.f, 0.f, 0.f, 0.f, 0.f, 0.f, 0.f, 0.f};
    v8f acc[MT];
#pragma unroll
    for (int mt = 0; mt < MT; ++mt) acc[mt] = zero;

    const int mrow = lane & 15;
    const int kb   = (lane & 16) ? 16 : 0;

#pragma unroll
    for (int k0 = 0; k0 < I; k0 += 32) {
        v16h bf = load_b_frag(Xb, NPOS, k0, n0, lane);
#pragma unroll
        for (int mt = 0; mt < MT; ++mt) {
            int m = mt * 16 + mrow;
            v16h af;
            if (m < O) {                     // always true unless O%16 != 0
                const _Float16* p = &Wl[m * I + k0 + kb];
#pragma unroll
                for (int e = 0; e < 8; ++e) af[e]     = p[e];       // K=kb+0..7
#pragma unroll
                for (int e = 0; e < 8; ++e) af[e + 8] = p[e + 8];   // K=kb+8..15
            } else {
#pragma unroll
                for (int e = 0; e < 16; ++e) af[e] = (_Float16)0.0f;
            }
            acc[mt] = __builtin_amdgcn_wmma_f32_16x16x32_f16(
                          false, af, false, bf, (short)0, acc[mt], false, false);
        }
    }

    // C/D layout: VGPR r -> M = r + 8*(lane>=16), N = lane&15
    int n     = n0 + (lane & 15);
    int rbase = (lane & 16) ? 8 : 0;
    float*       Yb = Y + (size_t)b * O * NPOS;
    const float* Rb = RESID ? (resid + (size_t)b * O * NPOS) : nullptr;
#pragma unroll
    for (int mt = 0; mt < MT; ++mt) {
#pragma unroll
        for (int r = 0; r < 8; ++r) {
            int o = mt * 16 + rbase + r;
            if (o < O) {
                float v = acc[mt][r];
                if (HAS_SCALE) v = sc[o] * v + bi[o];
                else           v = v + bi[o];
                if (RELU)  v = fmaxf(v, 0.0f);
                if (RESID) v += Rb[(size_t)o * NPOS + n];
                Yb[(size_t)o * NPOS + n] = v;
            }
        }
    }
}

// ---------------------------------------------------------------------------
// Depthwise 5x5x5, pad 2 (zero pad), fused BN affine.  A,Out: [B,32,NPOS]
// ---------------------------------------------------------------------------
__global__ __launch_bounds__(256)
void dw5_kernel(const float* __restrict__ A, const float* __restrict__ Wd,
                const float* __restrict__ s2, const float* __restrict__ b2,
                float* __restrict__ Out) {
    int idx = blockIdx.x * 256 + threadIdx.x;          // B*32*NPOS
    int n = idx & (NPOS - 1);
    int c = (idx >> 15) & 31;
    int b = idx >> 20;
    int w = n & 31, h = (n >> 5) & 31, d = n >> 10;
    const float* Ac = A + (((size_t)b * 32 + c) << 15);
    const float* wc = Wd + c * 125;
    float acc = 0.0f;
    for (int kd = 0; kd < 5; ++kd) {
        int dd = d + kd - 2;
        if ((unsigned)dd >= 32u) continue;
        for (int kh = 0; kh < 5; ++kh) {
            int hh = h + kh - 2;
            if ((unsigned)hh >= 32u) continue;
            int base = (dd << 10) + (hh << 5);
            const float* wrow = wc + kd * 25 + kh * 5;
#pragma unroll
            for (int kw = 0; kw < 5; ++kw) {
                int ww = w + kw - 2;
                if ((unsigned)ww < 32u) acc += Ac[base + ww] * wrow[kw];
            }
        }
    }
    Out[idx] = s2[c] * acc + b2[c];
}

// ---------------------------------------------------------------------------
// GroupNorm stats: per (b,g) region of 27*NPOS contiguous floats.
// 64 blocks per region, LDS tree reduce, 2 atomics per block.
// ---------------------------------------------------------------------------
__global__ __launch_bounds__(256)
void gn_reduce(const float* __restrict__ Wk, float* __restrict__ stats) {
    int region = blockIdx.x >> 6;      // b*8+g, 0..15
    int blk    = blockIdx.x & 63;
    const float* base = Wk + (size_t)region * 27 * NPOS;
    const int Ntot = 27 * NPOS;        // 884736
    float s = 0.f, ss = 0.f;
    for (int i = blk * 256 + threadIdx.x; i < Ntot; i += 64 * 256) {
        float v = base[i];
        s += v; ss += v * v;
    }
    __shared__ float sh[256], sh2[256];
    sh[threadIdx.x] = s; sh2[threadIdx.x] = ss;
    __syncthreads();
    for (int off = 128; off > 0; off >>= 1) {
        if (threadIdx.x < off) {
            sh[threadIdx.x]  += sh[threadIdx.x + off];
            sh2[threadIdx.x] += sh2[threadIdx.x + off];
        }
        __syncthreads();
    }
    if (threadIdx.x == 0) {
        atomicAdd(&stats[region * 2],     sh[0]);
        atomicAdd(&stats[region * 2 + 1], sh2[0]);
    }
}

__global__ void zero32(float* p) { if (threadIdx.x < 32) p[threadIdx.x] = 0.0f; }

__global__ void gn_finalize(const float* __restrict__ raw, float* __restrict__ out) {
    int i = threadIdx.x;
    if (i < 16) {
        float cnt = 27.0f * (float)NPOS;
        float mu  = raw[2 * i] / cnt;
        float var = raw[2 * i + 1] / cnt - mu * mu;
        out[2 * i]     = mu;
        out[2 * i + 1] = rsqrtf(var + GN_EPS);
    }
}

// ---------------------------------------------------------------------------
// SKA: y[b,c,n] = bn_s[c]*sum_k x_wrap[b,c,k,n]*wk_hat[b,g,k,n] + bn_b[c] + x[b,c,n]
// wk_hat = (wk_raw - mu)*rsig*gn_g + gn_b applied on the fly.
// One thread per (b, g, position); 8 channels per group reuse the 27 taps.
// ---------------------------------------------------------------------------
__global__ __launch_bounds__(256)
void ska_kernel(const float* __restrict__ X, const float* __restrict__ Wk,
                const float* __restrict__ st, const float* __restrict__ gng,
                const float* __restrict__ gnb, const float* __restrict__ bns,
                const float* __restrict__ bnb, float* __restrict__ Y) {
    int idx = blockIdx.x * 256 + threadIdx.x;        // B*8*NPOS
    int n = idx & (NPOS - 1);
    int g = (idx >> 15) & 7;
    int b = idx >> 18;
    int w = n & 31, h = (n >> 5) & 31, d = n >> 10;
    // roll(x, -di): patches[k][d] = x[(d+di) mod 32], di = a-1
    int dm[3] = { ((d + 31) & 31) << 10, d << 10, ((d + 1) & 31) << 10 };
    int hm[3] = { ((h + 31) & 31) << 5,  h << 5,  ((h + 1) & 31) << 5 };
    int wm[3] = { (w + 31) & 31,         w,       (w + 1) & 31 };

    float mu = st[(b * 8 + g) * 2];
    float rs = st[(b * 8 + g) * 2 + 1];
    const float* Xg = X  + (((size_t)b * 64  + g * 8 ) << 15);
    const float* Wg = Wk + (((size_t)b * 216 + g * 27) << 15) + n;

    float acc[8] = {0, 0, 0, 0, 0, 0, 0, 0};
    int k = 0;
    for (int a = 0; a < 3; ++a)
        for (int bb = 0; bb < 3; ++bb)
            for (int cc = 0; cc < 3; ++cc, ++k) {
                int ch   = g * 27 + k;
                float wv = (Wg[(size_t)k << 15] - mu) * rs * gng[ch] + gnb[ch];
                int np   = dm[a] + hm[bb] + wm[cc];
#pragma unroll
                for (int c8 = 0; c8 < 8; ++c8)
                    acc[c8] += Xg[((size_t)c8 << 15) + np] * wv;
            }
#pragma unroll
    for (int c8 = 0; c8 < 8; ++c8) {
        int c = g * 8 + c8;
        float xv = Xg[((size_t)c8 << 15) + n];
        Y[(((size_t)b * 64 + c) << 15) + n] = bns[c] * acc[c8] + bnb[c] + xv;
    }
}

// ---------------------------------------------------------------------------
// Launcher
// ---------------------------------------------------------------------------
extern "C" void kernel_launch(void* const* d_in, const int* in_sizes, int n_in,
                              void* d_out, int out_size, void* d_ws, size_t ws_size,
                              hipStream_t stream) {
    const float* x     = (const float*)d_in[0];
    const float* w1    = (const float*)d_in[1];
    const float* s1    = (const float*)d_in[2];
    const float* b1    = (const float*)d_in[3];
    const float* w2    = (const float*)d_in[4];
    const float* s2    = (const float*)d_in[5];
    const float* b2    = (const float*)d_in[6];
    const float* w3    = (const float*)d_in[7];
    const float* s3    = (const float*)d_in[8];
    const float* b3    = (const float*)d_in[9];
    const float* w4    = (const float*)d_in[10];
    const float* b4    = (const float*)d_in[11];
    const float* gn_g  = (const float*)d_in[12];
    const float* gn_b  = (const float*)d_in[13];
    const float* bn_s  = (const float*)d_in[14];
    const float* bn_b  = (const float*)d_in[15];
    const float* pw1_w = (const float*)d_in[16];
    const float* pw1_s = (const float*)d_in[17];
    const float* pw1_b = (const float*)d_in[18];
    const float* pw2_w = (const float*)d_in[19];
    const float* pw2_s = (const float*)d_in[20];
    const float* pw2_b = (const float*)d_in[21];
    float* out = (float*)d_out;

    // Workspace layout (floats), with aliasing to fit ~82 MB (all L2-resident):
    float* ws        = (float*)d_ws;
    float* a1        = ws;                 // 2*32*NPOS  = 2,097,152
    float* a2        = a1 + 2097152;       // 2*32*NPOS
    float* a3        = a2 + 2097152;       // 2*32*NPOS
    float* wk        = a3 + 2097152;       // 2*216*NPOS = 14,155,776
    float* stats_raw = wk + 14155776;      // 32
    float* stats_fin = stats_raw + 32;     // 32
    float* y  = ws;   // reuses a1+a2 region (exactly 2*64*NPOS floats; free by then)
    float* f1 = wk;   // reuses wk region   (2*128*NPOS <= 2*216*NPOS; free by then)

    // Column-strip GEMMs: waves = B * (NPOS/16) = 4096 -> 512 blocks of 8 waves
    // 1) cv1 + BN + ReLU  (WMMA GEMM 64->32)
    conv1x1_wmma<32, 64, true, true, false><<<512, 256, 0, stream>>>(
        x, w1, s1, b1, nullptr, a1);
    // 2) dw 5^3 + BN
    dw5_kernel<<<8192, 256, 0, stream>>>(a1, w2, s2, b2, a2);
    // 3) cv3 + BN + ReLU  (WMMA GEMM 32->32)
    conv1x1_wmma<32, 32, true, true, false><<<512, 256, 0, stream>>>(
        a2, w3, s3, b3, nullptr, a3);
    // 4) cv4 + bias       (WMMA GEMM 32->216)
    conv1x1_wmma<216, 32, false, false, false><<<512, 256, 0, stream>>>(
        a3, w4, nullptr, b4, nullptr, wk);
    // 5-7) GroupNorm statistics
    zero32<<<1, 32, 0, stream>>>(stats_raw);
    gn_reduce<<<1024, 256, 0, stream>>>(wk, stats_raw);
    gn_finalize<<<1, 32, 0, stream>>>(stats_raw, stats_fin);
    // 8) SKA + BN + residual
    ska_kernel<<<2048, 256, 0, stream>>>(
        x, wk, stats_fin, gn_g, gn_b, bn_s, bn_b, y);
    // 9) FFN pw1 + BN + ReLU (WMMA GEMM 64->128)
    conv1x1_wmma<128, 64, true, true, false><<<512, 256, 0, stream>>>(
        y, pw1_w, pw1_s, pw1_b, nullptr, f1);
    // 10) FFN pw2 + BN + residual y (WMMA GEMM 128->64) -> d_out
    conv1x1_wmma<64, 128, true, false, true><<<512, 256, 0, stream>>>(
        f1, pw2_w, pw2_s, pw2_b, y, out);
}
